// GAT_72490458021888
// MI455X (gfx1250) — matrix operand order repaired and checked by reading, not compiled, not run on hardware
//
#include <hip/hip_runtime.h>
#include <hip/hip_bf16.h>

#define GAT_HEADS 4
#define GAT_GRAPHS 64

typedef __attribute__((ext_vector_type(16))) _Float16 v16h;
typedef __attribute__((ext_vector_type(8)))  float    v8f;

// ---------- order-preserving float <-> uint key for atomic segment-max ----------
__device__ __forceinline__ unsigned gat_fkey(float f) {
    unsigned u = __float_as_uint(f);
    return (u & 0x80000000u) ? ~u : (u | 0x80000000u);
}
__device__ __forceinline__ float gat_fdecode(unsigned k) {
    return __uint_as_float((k & 0x80000000u) ? (k & 0x7fffffffu) : ~k);
}

// =======================================================================
// WMMA GEMM:  C[rows x NC] = A[rows x K] (f32) * B[K x NC] (f32)
//   - K = KSTEPS*32, fully unrolled -> 2*KSTEPS v_wmma_f32_16x16x32_f16
//   - Block = 256 threads (8 waves); block tile = 128 rows x 32 cols
//   - Wave tile = 16 rows x 32 cols (two accumulators share one A frag)
//   - B tile (32x32 f32, 4KB) staged to LDS with per-lane async DMA
//     (global_load_async_to_lds_b128, ASYNCcnt-tracked), f16 convert on read
// =======================================================================
template <int KSTEPS>
__global__ __launch_bounds__(256) void GAT_gemm_wmma(const float* __restrict__ A,
                                                     const float* __restrict__ B,
                                                     float* __restrict__ C,
                                                     int rows, int NC) {
    constexpr int K = KSTEPS * 32;
    __shared__ float BsF[32 * 32];             // one K-step of B: 32 x 32 (f32)

    const int lane    = threadIdx.x & 31;
    const int wave    = threadIdx.x >> 5;
    const int l15     = lane & 15;
    const int halfSel = lane >> 4;             // 0: lanes 0-15, 1: lanes 16-31
    const int rowTile = blockIdx.x * 128 + wave * 16;
    const int colTile = blockIdx.y * 32;

    const int myRow      = rowTile + l15;
    const int rowClamped = myRow < rows ? myRow : rows - 1;   // keep EXEC full

    // per-thread staging slot: thread t copies B[k0 + (t>>3)][colTile + (t&7)*4 .. +3]
    const int skk = threadIdx.x >> 3;          // 0..31
    const int snn = (threadIdx.x & 7) << 2;    // 0,4,...,28
    const unsigned ldsAddr = (unsigned)(size_t)(&BsF[skk * 32 + snn]);

    v8f acc0 = {};
    v8f acc1 = {};

#pragma unroll
    for (int s = 0; s < KSTEPS; ++s) {
        const int k0 = s * 32;

        __syncthreads();                       // previous-iter readers done
        {   // async DMA: 256 lanes x 16B = full 32x32 f32 tile
            const float* gsrc = B + (size_t)(k0 + skk) * NC + colTile + snn;
            asm volatile("global_load_async_to_lds_b128 %0, %1, off"
                         :: "v"(ldsAddr), "v"(gsrc)
                         : "memory");
        }
        asm volatile("s_wait_asynccnt 0x0" ::: "memory");
        __syncthreads();                       // tile visible to all waves

        // ---- A fragment: lane (l15,halfSel) -> K = k0 + halfSel*8 + {0..7,16..23}
        const float4* ap =
            (const float4*)(A + (size_t)rowClamped * K + k0 + halfSel * 8);
        float4 a0 = ap[0], a1 = ap[1], a2 = ap[4], a3 = ap[5];
        v16h a;
        a[0]  = (_Float16)a0.x; a[1]  = (_Float16)a0.y;
        a[2]  = (_Float16)a0.z; a[3]  = (_Float16)a0.w;
        a[4]  = (_Float16)a1.x; a[5]  = (_Float16)a1.y;
        a[6]  = (_Float16)a1.z; a[7]  = (_Float16)a1.w;
        a[8]  = (_Float16)a2.x; a[9]  = (_Float16)a2.y;
        a[10] = (_Float16)a2.z; a[11] = (_Float16)a2.w;
        a[12] = (_Float16)a3.x; a[13] = (_Float16)a3.y;
        a[14] = (_Float16)a3.z; a[15] = (_Float16)a3.w;

        // ---- B fragments: lane (l15,halfSel) -> K = halfSel*8 + {0..7,16..23}; col = n
        v16h b0, b1;
#pragma unroll
        for (int h = 0; h < 16; ++h) {
            int k = (h & 7) + (halfSel << 3) + ((h >> 3) << 4);
            b0[h] = (_Float16)BsF[k * 32 + l15];
            b1[h] = (_Float16)BsF[k * 32 + 16 + l15];
        }

        acc0 = __builtin_amdgcn_wmma_f32_16x16x32_f16(
                   false, a, false, b0, (short)0, acc0, false, false);
        acc1 = __builtin_amdgcn_wmma_f32_16x16x32_f16(
                   false, a, false, b1, (short)0, acc1, false, false);
    }

    // store: VGPR r -> M = r + halfSel*8, N = l15 (and +16 for acc1)
#pragma unroll
    for (int r = 0; r < 8; ++r) {
        int row = rowTile + r + halfSel * 8;
        if (row < rows) {
            C[(size_t)row * NC + colTile + l15]      = acc0[r];
            C[(size_t)row * NC + colTile + 16 + l15] = acc1[r];
        }
    }
}

// =======================================================================
// Per-node attention dot products: s[n,h] = <H[n,h,:], a_src[h,:]>, same for d
// =======================================================================
__global__ void GAT_scores(const float* __restrict__ H,
                           const float* __restrict__ a_src,
                           const float* __restrict__ a_dst,
                           float* __restrict__ s, float* __restrict__ d,
                           int N, int Cc) {
    int idx = blockIdx.x * blockDim.x + threadIdx.x;   // n*HEADS + head
    if (idx >= N * GAT_HEADS) return;
    int n = idx >> 2, hd = idx & 3;
    const float* hp = H + (size_t)n * (GAT_HEADS * Cc) + hd * Cc;
    const float* as = a_src + hd * Cc;
    const float* ad = a_dst + hd * Cc;
    float ss = 0.f, dd = 0.f;
    for (int c = 0; c < Cc; ++c) {
        float v = hp[c];
        ss += v * as[c];
        dd += v * ad[c];
    }
    s[idx] = ss;
    d[idx] = dd;
}

// =======================================================================
// Edge pass 1: e = leaky_relu(s[src]+d[dst], 0.2); atomic segment-max by dst
// =======================================================================
__global__ void GAT_edge1(const int* __restrict__ src, const int* __restrict__ dst,
                          const float* __restrict__ s, const float* __restrict__ d,
                          float* __restrict__ ev, unsigned* __restrict__ mkey,
                          int Eorig, int Etot) {
    int idx = blockIdx.x * blockDim.x + threadIdx.x;   // e*HEADS + head
    if (idx >= Etot * GAT_HEADS) return;
    int e = idx >> 2, h = idx & 3;
    int si, di;
    if (e < Eorig) { si = src[e]; di = dst[e]; }
    else           { si = di = e - Eorig; }            // appended self-loops
    float v = s[si * GAT_HEADS + h] + d[di * GAT_HEADS + h];
    v = v > 0.f ? v : 0.2f * v;
    ev[idx] = v;
    atomicMax(&mkey[di * GAT_HEADS + h], gat_fkey(v));
}

// =======================================================================
// Edge pass 2: ex = exp(e - m[dst]); denom[dst] += ex (segment softmax denom)
// =======================================================================
__global__ void GAT_edge2(const int* __restrict__ dst,
                          float* __restrict__ ev,
                          const unsigned* __restrict__ mkey,
                          float* __restrict__ denom,
                          int Eorig, int Etot) {
    int idx = blockIdx.x * blockDim.x + threadIdx.x;
    if (idx >= Etot * GAT_HEADS) return;
    int e = idx >> 2, h = idx & 3;
    int di = (e < Eorig) ? dst[e] : (e - Eorig);
    float m  = gat_fdecode(mkey[di * GAT_HEADS + h]);
    float ex = __expf(ev[idx] - m);
    ev[idx] = ex;                                      // overwrite with exp term
    atomicAdd(&denom[di * GAT_HEADS + h], ex);
}

// =======================================================================
// Edge aggregation: out[dst, c] += h[src, c] * alpha(e, head(c))
// One thread handles 4 consecutive channels of one edge (float4 gather).
// =======================================================================
__global__ void GAT_aggregate(const int* __restrict__ src, const int* __restrict__ dst,
                              const float* __restrict__ H,
                              const float* __restrict__ ex,
                              const float* __restrict__ denom,
                              float* __restrict__ out,
                              int Eorig, int Etot, int HC, int Cc) {
    int chunks = HC >> 2;
    int idx = blockIdx.x * blockDim.x + threadIdx.x;
    if (idx >= Etot * chunks) return;
    int e   = idx / chunks;
    int c   = (idx - e * chunks) << 2;
    int hd  = c / Cc;
    int si, di;
    if (e < Eorig) { si = src[e]; di = dst[e]; }
    else           { si = di = e - Eorig; }
    float alpha = ex[e * GAT_HEADS + hd] / (denom[di * GAT_HEADS + hd] + 1e-16f);
    const float4* hp = (const float4*)(H + (size_t)si * HC + c);
    __builtin_prefetch(hp, 0, 1);                      // global_prefetch_b8
    float4 hv = *hp;
    float* op = out + (size_t)di * HC + c;
    atomicAdd(op + 0, hv.x * alpha);
    atomicAdd(op + 1, hv.y * alpha);
    atomicAdd(op + 2, hv.z * alpha);
    atomicAdd(op + 3, hv.w * alpha);
}

// =======================================================================
// out = relu(out + bias)  (bias applied after aggregation, PyG semantics)
// =======================================================================
__global__ void GAT_bias_relu(float* __restrict__ out, const float* __restrict__ b,
                              int total, int HC) {
    int idx = blockIdx.x * blockDim.x + threadIdx.x;
    if (idx >= total) return;
    float v = out[idx] + b[idx % HC];
    out[idx] = v > 0.f ? v : 0.f;
}

// =======================================================================
// Global mean-pool accumulation per graph
// =======================================================================
__global__ void GAT_pool(const float* __restrict__ h, const int* __restrict__ batch,
                         float* __restrict__ sums, float* __restrict__ counts,
                         int N, int HC) {
    int chunks = HC >> 2;
    int idx = blockIdx.x * blockDim.x + threadIdx.x;
    if (idx >= N * chunks) return;
    int n = idx / chunks;
    int c = (idx - n * chunks) << 2;
    int g = batch[n];
    float4 v = *(const float4*)(h + (size_t)n * HC + c);
    float* sp = sums + (size_t)g * HC + c;
    atomicAdd(sp + 0, v.x);
    atomicAdd(sp + 1, v.y);
    atomicAdd(sp + 2, v.z);
    atomicAdd(sp + 3, v.w);
    if (c == 0) atomicAdd(&counts[g], 1.0f);
}

// =======================================================================
// Head: pooled = sums/max(count,1); logits = pooled @ Wout + bout; softmax(2)
// =======================================================================
__global__ void GAT_head(const float* __restrict__ sums, const float* __restrict__ counts,
                         const float* __restrict__ Wout, const float* __restrict__ bout,
                         float* __restrict__ out, int HC) {
    int g = threadIdx.x;
    if (g >= GAT_GRAPHS) return;
    float cnt = counts[g];
    cnt = cnt > 1.0f ? cnt : 1.0f;
    float l0 = bout[0], l1 = bout[1];
    const float* sp = sums + (size_t)g * HC;
    for (int c = 0; c < HC; ++c) {
        float p = sp[c] / cnt;
        l0 += p * Wout[c * 2 + 0];
        l1 += p * Wout[c * 2 + 1];
    }
    float m  = l0 > l1 ? l0 : l1;
    float e0 = __expf(l0 - m), e1 = __expf(l1 - m);
    float inv = 1.0f / (e0 + e1);
    out[g * 2 + 0] = e0 * inv;
    out[g * 2 + 1] = e1 * inv;
}

// =======================================================================
extern "C" void kernel_launch(void* const* d_in, const int* in_sizes, int n_in,
                              void* d_out, int out_size, void* d_ws, size_t ws_size,
                              hipStream_t stream) {
    const float* x      = (const float*)d_in[0];
    const int*   ei     = (const int*)d_in[1];     // [2, E] concatenated
    const int*   batch  = (const int*)d_in[2];
    const float* W1     = (const float*)d_in[3];
    const float* a_src1 = (const float*)d_in[4];
    const float* a_dst1 = (const float*)d_in[5];
    const float* b1     = (const float*)d_in[6];
    const float* W2     = (const float*)d_in[7];
    const float* a_src2 = (const float*)d_in[8];
    const float* a_dst2 = (const float*)d_in[9];
    const float* b2     = (const float*)d_in[10];
    const float* Wout   = (const float*)d_in[11];
    const float* bout   = (const float*)d_in[12];

    const int IN_CH = 128, C1 = 64, C2 = 32;
    const int HC1 = GAT_HEADS * C1;                // 256
    const int HC2 = GAT_HEADS * C2;                // 128
    const int N     = in_sizes[0] / IN_CH;         // 50000
    const int Eorig = in_sizes[1] / 2;             // 800000
    const int Etot  = Eorig + N;                   // + self-loops
    const int* src  = ei;
    const int* dstp = ei + Eorig;

    // ---------------- workspace layout ----------------
    char*  ws  = (char*)d_ws;
    size_t off = 0;
    auto take = [&](size_t nfloats) -> float* {
        float* p = (float*)(ws + off);
        off += nfloats * sizeof(float);
        return p;
    };
    // zero-initialized region (one contiguous memset)
    float*    out1   = take((size_t)N * HC1);
    float*    out2   = take((size_t)N * HC2);
    unsigned* mkey1  = (unsigned*)take((size_t)N * GAT_HEADS);
    float*    den1   = take((size_t)N * GAT_HEADS);
    unsigned* mkey2  = (unsigned*)take((size_t)N * GAT_HEADS);
    float*    den2   = take((size_t)N * GAT_HEADS);
    float*    sums   = take((size_t)GAT_GRAPHS * HC2);
    float*    counts = take((size_t)GAT_GRAPHS);
    size_t zeroBytes = off;
    // uninitialized scratch
    float* H1    = take((size_t)N * HC1);
    float* H2    = take((size_t)N * HC2);
    float* s1    = take((size_t)N * GAT_HEADS);
    float* d1v   = take((size_t)N * GAT_HEADS);
    float* s2    = take((size_t)N * GAT_HEADS);
    float* d2v   = take((size_t)N * GAT_HEADS);
    float* exbuf = take((size_t)Etot * GAT_HEADS);
    (void)ws_size; (void)n_in; (void)out_size;

    hipMemsetAsync(d_ws, 0, zeroBytes, stream);

    const int B = 256;
    int e4 = Etot * GAT_HEADS;

    // ---------------- layer 1 (K=128 -> 4 K-steps) ----------------
    GAT_gemm_wmma<4><<<dim3((N + 127) / 128, HC1 / 32), B, 0, stream>>>(x, W1, H1, N, HC1);
    GAT_scores<<<(N * GAT_HEADS + B - 1) / B, B, 0, stream>>>(H1, a_src1, a_dst1, s1, d1v, N, C1);
    GAT_edge1<<<(e4 + B - 1) / B, B, 0, stream>>>(src, dstp, s1, d1v, exbuf, mkey1, Eorig, Etot);
    GAT_edge2<<<(e4 + B - 1) / B, B, 0, stream>>>(dstp, exbuf, mkey1, den1, Eorig, Etot);
    {
        int tot = Etot * (HC1 >> 2);
        GAT_aggregate<<<(tot + B - 1) / B, B, 0, stream>>>(src, dstp, H1, exbuf, den1, out1,
                                                           Eorig, Etot, HC1, C1);
    }
    GAT_bias_relu<<<((N * HC1) + B - 1) / B, B, 0, stream>>>(out1, b1, N * HC1, HC1);

    // ---------------- layer 2 (K=256 -> 8 K-steps) ----------------
    GAT_gemm_wmma<8><<<dim3((N + 127) / 128, HC2 / 32), B, 0, stream>>>(out1, W2, H2, N, HC2);
    GAT_scores<<<(N * GAT_HEADS + B - 1) / B, B, 0, stream>>>(H2, a_src2, a_dst2, s2, d2v, N, C2);
    GAT_edge1<<<(e4 + B - 1) / B, B, 0, stream>>>(src, dstp, s2, d2v, exbuf, mkey2, Eorig, Etot);
    GAT_edge2<<<(e4 + B - 1) / B, B, 0, stream>>>(dstp, exbuf, mkey2, den2, Eorig, Etot);
    {
        int tot = Etot * (HC2 >> 2);
        GAT_aggregate<<<(tot + B - 1) / B, B, 0, stream>>>(src, dstp, H2, exbuf, den2, out2,
                                                           Eorig, Etot, HC2, C2);
    }
    GAT_bias_relu<<<((N * HC2) + B - 1) / B, B, 0, stream>>>(out2, b2, N * HC2, HC2);

    // ---------------- pooling + head ----------------
    GAT_pool<<<((N * (HC2 >> 2)) + B - 1) / B, B, 0, stream>>>(out2, batch, sums, counts, N, HC2);
    GAT_head<<<1, GAT_GRAPHS, 0, stream>>>(sums, counts, Wout, bout, (float*)d_out, HC2);
}
